// MaskedAttention_11845519802435
// MI455X (gfx1250) — matrix operand order repaired and checked by reading, not compiled, not run on hardware
//
#include <hip/hip_runtime.h>
#include <hip/hip_bf16.h>

// ---------------------------------------------------------------------------
// MaskedAttention (B=8, T=C=2048) as six GEMM-NT passes + row softmax.
// All GEMMs are D = A * B^T with row-major A (M,K) and B (N,K):
//   k = x Wk^T + bk ; q = x Wq^T + bq ; s = k q^T ; p = softmax(s)
//   v = p Wv^T + bv ; y = p v^T ; out = y Wo^T + bo
// Compute-bound => bf16 inputs + v_wmma_f32_16x16x32_bf16 (fp32 accumulate),
// async global->LDS DMA (ASYNCcnt) with a double-buffered LDS pipeline.
// ---------------------------------------------------------------------------

typedef __bf16 bf16;
typedef __attribute__((ext_vector_type(16))) __bf16 v16bf;
typedef __attribute__((ext_vector_type(8)))  __bf16 v8bf;
typedef __attribute__((ext_vector_type(4)))  __bf16 v4bf;
typedef __attribute__((ext_vector_type(8)))  float   v8f;

#define BM 128
#define BN 128
#define BK 32
#define LDSS (BK + 8)   // padded bf16 stride (80B rows -> 16B aligned chunks)

#if __has_builtin(__builtin_amdgcn_global_load_async_to_lds_b128) && \
    __has_builtin(__builtin_amdgcn_s_wait_asynccnt)
#define HAVE_ASYNC_LDS 1
// builtin signature (from clang diagnostic): (i32x4 AS1*, i32x4 AS3*, Imm, Imm)
typedef int i32x4 __attribute__((vector_size(16)));
typedef __attribute__((address_space(1))) i32x4* gb128_t;
typedef __attribute__((address_space(3))) i32x4* lb128_t;
#else
#define HAVE_ASYNC_LDS 0
#endif

// ---------------- fp32 -> bf16 bulk conversion (vectorized x4) -------------
__global__ __launch_bounds__(256)
void cvt_f32_to_bf16(const float* __restrict__ in, bf16* __restrict__ out, long n4) {
  long i = (long)blockIdx.x * 256 + threadIdx.x;
  if (i >= n4) return;
  float4 v = ((const float4*)in)[i];
  v4bf o;
  o[0] = (bf16)v.x; o[1] = (bf16)v.y; o[2] = (bf16)v.z; o[3] = (bf16)v.w;
  ((v4bf*)out)[i] = o;
}

// ---------------- WMMA GEMM-NT: D[b] = A[b] (M,K) * Bw[b] (N,K)^T (+bias) --
// 256 threads = 8 waves; block tile 128x128x32; wave tile 32x64 (2x4 WMMAs).
template <int OUT_BF16, int HAS_BIAS>
__global__ __launch_bounds__(256)
void gemm_nt_wmma(const bf16* __restrict__ A, const bf16* __restrict__ Bw,
                  const float* __restrict__ bias, void* __restrict__ Dp,
                  int M, int N, int K, long sA, long sB, long sD)
{
  __shared__ bf16 As[2][BM][LDSS];
  __shared__ bf16 Bs[2][BN][LDSS];

  const int  tid   = threadIdx.x;
  const int  lane  = tid & 31;
  const int  wave  = tid >> 5;
  const int  waveM = (wave & 3) * 32;   // 4 waves along M (128)
  const int  waveN = (wave >> 2) * 64;  // 2 waves along N (128)
  const int  lrow  = lane & 15;
  const int  khalf = lane >> 4;         // which K-half this lane holds

  const long gM0 = (long)blockIdx.y * BM;
  const long gN0 = (long)blockIdx.x * BN;
  const long bz  = blockIdx.z;

  const bf16* __restrict__ Ab = A  + bz * sA;
  const bf16* __restrict__ Bb = Bw + bz * sB;

  // per-thread staging chunks: A = 512 x 16B, B = 512 x 16B over 256 threads
  const int sr = tid >> 2;            // rows tid/4 and tid/4+64
  const int sc = (tid & 3) * 8;       // 8-bf16 column chunk

  v8f acc[2][4] = {};
  const int ntk = K / BK;

#if HAVE_ASYNC_LDS
  // ---- async DMA staging: global -> LDS directly, tracked by ASYNCcnt -----
  #define STAGE_TILE(t, buf)                                                   \
    {                                                                          \
      long k0 = (long)(t) * BK;                                                \
      __builtin_amdgcn_global_load_async_to_lds_b128(                          \
          (gb128_t)(Ab + (gM0 + sr) * (long)K + k0 + sc),                      \
          (lb128_t)(&As[buf][sr][sc]), 0, 0);                                  \
      __builtin_amdgcn_global_load_async_to_lds_b128(                          \
          (gb128_t)(Ab + (gM0 + sr + 64) * (long)K + k0 + sc),                 \
          (lb128_t)(&As[buf][sr + 64][sc]), 0, 0);                             \
      __builtin_amdgcn_global_load_async_to_lds_b128(                          \
          (gb128_t)(Bb + (gN0 + sr) * (long)K + k0 + sc),                      \
          (lb128_t)(&Bs[buf][sr][sc]), 0, 0);                                  \
      __builtin_amdgcn_global_load_async_to_lds_b128(                          \
          (gb128_t)(Bb + (gN0 + sr + 64) * (long)K + k0 + sc),                 \
          (lb128_t)(&Bs[buf][sr + 64][sc]), 0, 0);                             \
    }

  STAGE_TILE(0, 0)
  for (int t = 0; t < ntk; ++t) {
    const int cur = t & 1;
    __builtin_amdgcn_s_wait_asynccnt(0);  // own tile-t DMA complete
    __syncthreads();                      // all waves' tile-t in LDS; t-1 consumed
    if (t + 1 < ntk) STAGE_TILE(t + 1, cur ^ 1)  // DMA overlaps WMMAs below

    union U { v16bf v; v8bf h[2]; };
    U a[2], b[4];
#pragma unroll
    for (int mt = 0; mt < 2; ++mt) {
      const bf16* p = &As[cur][waveM + mt * 16 + lrow][khalf * 16];
      a[mt].h[0] = *(const v8bf*)(p);
      a[mt].h[1] = *(const v8bf*)(p + 8);
    }
#pragma unroll
    for (int nt = 0; nt < 4; ++nt) {
      const bf16* p = &Bs[cur][waveN + nt * 16 + lrow][khalf * 16];
      b[nt].h[0] = *(const v8bf*)(p);
      b[nt].h[1] = *(const v8bf*)(p + 8);
    }
#pragma unroll
    for (int mt = 0; mt < 2; ++mt)
#pragma unroll
      for (int nt = 0; nt < 4; ++nt)
        acc[mt][nt] = __builtin_amdgcn_wmma_f32_16x16x32_bf16(
            false, a[mt].v, false, b[nt].v, (short)0, acc[mt][nt], false, false);
  }
  #undef STAGE_TILE
#else
  // ---- fallback: global -> VGPR -> LDS, double-buffered, 2 barriers/iter --
  for (int t = 0; t < ntk; ++t) {
    const int cur = t & 1;
    long k0 = (long)t * BK;
    v8bf ra0 = *(const v8bf*)(Ab + (gM0 + sr) * (long)K + k0 + sc);
    v8bf ra1 = *(const v8bf*)(Ab + (gM0 + sr + 64) * (long)K + k0 + sc);
    v8bf rb0 = *(const v8bf*)(Bb + (gN0 + sr) * (long)K + k0 + sc);
    v8bf rb1 = *(const v8bf*)(Bb + (gN0 + sr + 64) * (long)K + k0 + sc);
    __syncthreads();
    *(v8bf*)(&As[cur][sr][sc])      = ra0;
    *(v8bf*)(&As[cur][sr + 64][sc]) = ra1;
    *(v8bf*)(&Bs[cur][sr][sc])      = rb0;
    *(v8bf*)(&Bs[cur][sr + 64][sc]) = rb1;
    if (t + 1 < ntk) {
      __builtin_prefetch(Ab + (gM0 + sr) * (long)K + k0 + BK + sc, 0, 3);
      __builtin_prefetch(Bb + (gN0 + sr) * (long)K + k0 + BK + sc, 0, 3);
    }
    __syncthreads();

    union U { v16bf v; v8bf h[2]; };
    U a[2], b[4];
#pragma unroll
    for (int mt = 0; mt < 2; ++mt) {
      const bf16* p = &As[cur][waveM + mt * 16 + lrow][khalf * 16];
      a[mt].h[0] = *(const v8bf*)(p);
      a[mt].h[1] = *(const v8bf*)(p + 8);
    }
#pragma unroll
    for (int nt = 0; nt < 4; ++nt) {
      const bf16* p = &Bs[cur][waveN + nt * 16 + lrow][khalf * 16];
      b[nt].h[0] = *(const v8bf*)(p);
      b[nt].h[1] = *(const v8bf*)(p + 8);
    }
#pragma unroll
    for (int mt = 0; mt < 2; ++mt)
#pragma unroll
      for (int nt = 0; nt < 4; ++nt)
        acc[mt][nt] = __builtin_amdgcn_wmma_f32_16x16x32_bf16(
            false, a[mt].v, false, b[nt].v, (short)0, acc[mt][nt], false, false);
  }
#endif

  // ---- epilogue: C/D layout VGPR i -> M = i (lanes<16) or i+8 (lanes>=16) -
  float* Df = (float*)Dp;
  bf16*  Db = (bf16*)Dp;
#pragma unroll
  for (int mt = 0; mt < 2; ++mt) {
#pragma unroll
    for (int nt = 0; nt < 4; ++nt) {
      long col  = gN0 + waveN + nt * 16 + lrow;
      long row0 = gM0 + waveM + mt * 16 + khalf * 8;
      float bv = HAS_BIAS ? bias[col] : 0.0f;
#pragma unroll
      for (int i = 0; i < 8; ++i) {
        float v = acc[mt][nt][i] + bv;
        long off = bz * sD + (row0 + i) * (long)N + col;
        if (OUT_BF16) Db[off] = (bf16)v;
        else          Df[off] = v;
      }
    }
  }
}

// ---------------- row softmax: f32 scores -> bf16 probs, 2048 cols ---------
__global__ __launch_bounds__(256)
void softmax_row_2048(const float* __restrict__ S, bf16* __restrict__ P) {
  const long row = blockIdx.x;
  const float* __restrict__ x = S + row * 2048L;
  bf16* __restrict__ y = P + row * 2048L;
  const int tid = threadIdx.x, lane = tid & 31, wave = tid >> 5;

  float vals[8], m = -3.0e38f;
#pragma unroll
  for (int i = 0; i < 8; ++i) { vals[i] = x[tid + i * 256]; m = fmaxf(m, vals[i]); }

  __shared__ float red[8];
#pragma unroll
  for (int off = 16; off > 0; off >>= 1) m = fmaxf(m, __shfl_xor(m, off, 32));
  if (lane == 0) red[wave] = m;
  __syncthreads();
  float mall = red[0];
#pragma unroll
  for (int w = 1; w < 8; ++w) mall = fmaxf(mall, red[w]);
  __syncthreads();

  float e[8], s = 0.0f;
#pragma unroll
  for (int i = 0; i < 8; ++i) { e[i] = __expf(vals[i] - mall); s += e[i]; }
#pragma unroll
  for (int off = 16; off > 0; off >>= 1) s += __shfl_xor(s, off, 32);
  if (lane == 0) red[wave] = s;
  __syncthreads();
  float sall = 0.0f;
#pragma unroll
  for (int w = 0; w < 8; ++w) sall += red[w];
  float inv = 1.0f / sall;
#pragma unroll
  for (int i = 0; i < 8; ++i) y[tid + i * 256] = (bf16)(e[i] * inv);
}

// ---------------------------------------------------------------------------
extern "C" void kernel_launch(void* const* d_in, const int* in_sizes, int n_in,
                              void* d_out, int out_size, void* d_ws, size_t ws_size,
                              hipStream_t stream) {
  const long Bb = 8, T = 2048, C = 2048;

  const float* x  = (const float*)d_in[0];
  const float* Wk = (const float*)d_in[1];
  const float* bk = (const float*)d_in[2];
  const float* Wq = (const float*)d_in[3];
  const float* bq = (const float*)d_in[4];
  const float* Wv = (const float*)d_in[5];
  const float* bv = (const float*)d_in[6];
  const float* Wo = (const float*)d_in[7];
  const float* bo = (const float*)d_in[8];

  // ---- workspace carve-out (256B aligned) ----
  size_t off = 0;
  char* base = (char*)d_ws;
  auto carve = [&](size_t bytes) -> char* {
    char* p = base + off;
    off += (bytes + 255) & ~(size_t)255;
    return p;
  };
  bf16*  xb  = (bf16*)carve(Bb * T * C * 2);   // 64 MiB
  bf16*  wkb = (bf16*)carve(C * C * 2);        // 8 MiB
  bf16*  wqb = (bf16*)carve(C * C * 2);
  bf16*  wvb = (bf16*)carve(C * C * 2);
  bf16*  wob = (bf16*)carve(C * C * 2);
  bf16*  kb  = (bf16*)carve(Bb * T * C * 2);   // 64 MiB (reused for v later)
  bf16*  qb  = (bf16*)carve(Bb * T * C * 2);   // 64 MiB (reused for y later)
  float* sc  = (float*)carve(Bb * T * T * 4);  // 128 MiB
  bf16*  pb  = (bf16*)carve(Bb * T * T * 2);   // 64 MiB
  bf16*  vb  = kb;  // k no longer needed once scores exist (stream-ordered)
  bf16*  yb  = qb;  // q no longer needed once scores exist

  // ---- 1) convert inputs to bf16 ----
  {
    long n4 = (Bb * T * C) / 4;
    cvt_f32_to_bf16<<<dim3((unsigned)((n4 + 255) / 256)), 256, 0, stream>>>(x, xb, n4);
    long w4 = (C * C) / 4;
    dim3 g((unsigned)((w4 + 255) / 256));
    cvt_f32_to_bf16<<<g, 256, 0, stream>>>(Wk, wkb, w4);
    cvt_f32_to_bf16<<<g, 256, 0, stream>>>(Wq, wqb, w4);
    cvt_f32_to_bf16<<<g, 256, 0, stream>>>(Wv, wvb, w4);
    cvt_f32_to_bf16<<<g, 256, 0, stream>>>(Wo, wob, w4);
  }

  const dim3 blk(256);
  const dim3 gFlat((unsigned)(C / BN), (unsigned)((Bb * T) / BM), 1);     // 16 x 128
  const dim3 gBat ((unsigned)(T / BN), (unsigned)(T / BM), (unsigned)Bb); // 16 x 16 x 8

  // ---- 2) k = x Wk^T + bk ; q = x Wq^T + bq (M = B*T flat) ----
  gemm_nt_wmma<1, 1><<<gFlat, blk, 0, stream>>>(xb, wkb, bk, kb,
      (int)(Bb * T), (int)C, (int)C, 0L, 0L, 0L);
  gemm_nt_wmma<1, 1><<<gFlat, blk, 0, stream>>>(xb, wqb, bq, qb,
      (int)(Bb * T), (int)C, (int)C, 0L, 0L, 0L);

  // ---- 3) scores[b] = k[b] q[b]^T  (batched NT, fp32 out) ----
  gemm_nt_wmma<0, 0><<<gBat, blk, 0, stream>>>(kb, qb, nullptr, sc,
      (int)T, (int)T, (int)C, T * C, T * C, T * T);

  // ---- 4) p = softmax(scores) -> bf16 ----
  softmax_row_2048<<<dim3((unsigned)(Bb * T)), blk, 0, stream>>>(sc, pb);

  // ---- 5) v = p Wv^T + bv (M = B*T flat) ----
  gemm_nt_wmma<1, 1><<<gFlat, blk, 0, stream>>>(pb, wvb, bv, vb,
      (int)(Bb * T), (int)C, (int)T, 0L, 0L, 0L);

  // ---- 6) y[b] = p[b] v[b]^T (batched NT) ----
  gemm_nt_wmma<1, 0><<<gBat, blk, 0, stream>>>(pb, vb, nullptr, yb,
      (int)T, (int)T, (int)T, T * T, T * C, T * T);

  // ---- 7) out = y Wo^T + bo (fp32 out to d_out) ----
  gemm_nt_wmma<0, 1><<<gFlat, blk, 0, stream>>>(yb, wob, bo, d_out,
      (int)(Bb * T), (int)C, (int)T, 0L, 0L, 0L);
}
